// MambaLayer_17841294147829
// MI455X (gfx1250) — matrix operand order repaired and checked by reading, not compiled, not run on hardware
//
#include <hip/hip_runtime.h>
#include <hip/hip_fp16.h>
#include <stdint.h>

#define BATCH   4
#define SEQ     2048
#define DMODEL  512
#define DSTATE  16
#define DCONV   4
#define DINNER  1024
#define DTRANK  32
#define NTOK    (BATCH*SEQ)        // 8192 tokens

typedef __attribute__((ext_vector_type(16))) _Float16 v16h;
typedef __attribute__((ext_vector_type(8)))  _Float16 v8h;
typedef __attribute__((ext_vector_type(8)))  float    v8f;

// ---------------------------------------------------------------- utilities
__global__ __launch_bounds__(256) void k_f32_to_f16(const float* __restrict__ src,
                                                    _Float16* __restrict__ dst, int n) {
    int i = blockIdx.x * 256 + threadIdx.x;
    if (i < n) dst[i] = (_Float16)src[i];
}

// ---------------------------------------------------------------- layernorm
// One wave (32 lanes) per token row of 512; 8 rows per 256-thread block.
__global__ __launch_bounds__(256) void k_layernorm(const float* __restrict__ x,
                                                   const float* __restrict__ gamma,
                                                   const float* __restrict__ beta,
                                                   _Float16* __restrict__ out) {
    int wave = threadIdx.x >> 5;
    int lane = threadIdx.x & 31;
    int row  = blockIdx.x * 8 + wave;
    const float* xr = x + (size_t)row * DMODEL;

    float v[16];
    float s = 0.f;
#pragma unroll
    for (int i = 0; i < 16; ++i) { v[i] = xr[lane + 32 * i]; s += v[i]; }
#pragma unroll
    for (int off = 16; off; off >>= 1) s += __shfl_xor(s, off, 32);
    float mu = s * (1.0f / DMODEL);

    float q = 0.f;
#pragma unroll
    for (int i = 0; i < 16; ++i) { float d = v[i] - mu; q += d * d; }
#pragma unroll
    for (int off = 16; off; off >>= 1) q += __shfl_xor(q, off, 32);
    float rstd = rsqrtf(q * (1.0f / DMODEL) + 1e-5f);

    _Float16* orow = out + (size_t)row * DMODEL;
#pragma unroll
    for (int i = 0; i < 16; ++i) {
        int c = lane + 32 * i;
        orow[c] = (_Float16)(((v[i] - mu) * rstd) * gamma[c] + beta[c]);
    }
}

// ---------------------------------------------------------------- WMMA GEMM
// C[M,N] = A[M,K] * W[N,K]^T  (A,W f16 row-major, C f32 row-major).
// Block = 8 waves, block tile 128(M) x 64(N); wave tile 16(M) x 64(N).
// The shared 64xK weight tile is staged through LDS with async copies,
// double-buffered: one global_load_async_to_lds_b128 per lane fills a
// 64x32-half (4 KB) stage while the previous stage feeds the WMMAs.
// All B fragments are loaded from LDS before the WMMA sequence so the
// scheduler can overlap ds latency with WMMA issue (partial dscnt waits).
// mode 0: plain store.  mode 1: softplus(acc + bias[n]) (dt_proj epilogue).
__global__ __launch_bounds__(256) void k_wmma_gemm(const _Float16* __restrict__ A,
                                                   const _Float16* __restrict__ W,
                                                   float* __restrict__ C,
                                                   int M, int N, int K,
                                                   int mode, const float* __restrict__ bias) {
    __shared__ _Float16 wtile[2][64 * 32];   // [stage][n_local*32 + k_local]

    int tid  = threadIdx.x;
    int wave = tid >> 5;
    int lane = tid & 31;
    int m0 = blockIdx.y * 128 + wave * 16;
    int n0 = blockIdx.x * 64;
    int hsel = lane >> 4;           // lane half: 0 -> lanes 0-15, 1 -> 16-31
    int lrow = lane & 15;

    // --- async staging geometry: 256 lanes x 16 B = 4 KB tile per stage
    int srow   = tid >> 2;          // 0..63 : tile row (n_local)
    int schunk = tid & 3;           // 0..3  : 16-byte chunk within 64-byte row
    const char* wsrc = (const char*)(W + (size_t)(n0 + srow) * K) + schunk * 16;
    unsigned lds_row_off = (unsigned)(srow * 64 + schunk * 16);
    unsigned lds_stage[2];
    lds_stage[0] = (unsigned)(uintptr_t)(&wtile[0][0]) + lds_row_off;
    lds_stage[1] = (unsigned)(uintptr_t)(&wtile[1][0]) + lds_row_off;

    // prologue: stage 0 <- K-slab 0
    {
        uint64_t g = (uint64_t)(uintptr_t)wsrc;
        asm volatile("global_load_async_to_lds_b128 %0, %1, off"
                     :: "v"(lds_stage[0]), "v"(g) : "memory");
    }

    v8f acc[4] = {};

    // A fragment base (CDNA5 16-bit A 16x32 layout):
    //  lanes 0-15 hold K {0..7, 16..23}; lanes 16-31 hold K {8..15, 24..31}
    const _Float16* arow = A + (size_t)(m0 + lrow) * K + (hsel ? 8 : 0);

    for (int k0 = 0, it = 0; k0 < K; k0 += 32, ++it) {
        int cur = it & 1;

        // make current stage visible to all waves (own asynccnt + barrier);
        // this barrier also guarantees the buffer we are about to overwrite
        // below has been fully consumed by every wave in the previous step.
        asm volatile("s_wait_asynccnt 0" ::: "memory");
        __syncthreads();

        if (k0 + 32 < K) {      // kick off next K-slab into the other stage
            uint64_t g = (uint64_t)(uintptr_t)(wsrc + (size_t)(k0 + 32) * 2);
            asm volatile("global_load_async_to_lds_b128 %0, %1, off"
                         :: "v"(lds_stage[cur ^ 1]), "v"(g) : "memory");
        }

        // A fragment from global (per-wave private rows)
        v8h alo = *(const v8h*)(arow + k0);
        v8h ahi = *(const v8h*)(arow + k0 + 16);
        v16h af;
#pragma unroll
        for (int i = 0; i < 8; ++i) { af[i] = alo[i]; af[i + 8] = ahi[i]; }

        // Load ALL four B fragments from LDS first (8x ds_load_b128 issued
        // back-to-back), then run the WMMA chain: lets the scheduler use
        // partial dscnt waits instead of a full stall per WMMA.
        v16h bf[4];
#pragma unroll
        for (int j = 0; j < 4; ++j) {
            const _Float16* bp = &wtile[cur][(j * 16 + lrow) * 32 + hsel * 8];
            v8h blo = *(const v8h*)(bp);
            v8h bhi = *(const v8h*)(bp + 16);
#pragma unroll
            for (int i = 0; i < 8; ++i) { bf[j][i] = blo[i]; bf[j][i + 8] = bhi[i]; }
        }

#pragma unroll
        for (int j = 0; j < 4; ++j) {
            acc[j] = __builtin_amdgcn_wmma_f32_16x16x32_f16(
                false, af, false, bf[j], (short)0, acc[j], false, false);
        }
    }

#pragma unroll
    for (int j = 0; j < 4; ++j) {
        int col = n0 + j * 16 + lrow;
        float bv = (mode == 1) ? bias[col] : 0.f;
#pragma unroll
        for (int r = 0; r < 8; ++r) {
            int row = m0 + r + hsel * 8;   // CDNA5 f32 C/D layout
            float v = acc[j][r];
            if (mode == 1) {               // softplus(acc + bias)
                v += bv;
                v = (v > 20.f) ? v : log1pf(__expf(v));
            }
            C[(size_t)row * N + col] = v;
        }
    }
}

// ------------------------------------------------- depthwise causal conv1d + SiLU
__global__ __launch_bounds__(256) void k_conv_silu(const float* __restrict__ xz,
                                                   const float* __restrict__ cw,
                                                   const float* __restrict__ cb,
                                                   float* __restrict__ xconv,
                                                   _Float16* __restrict__ xconv_h) {
    int idx = blockIdx.x * 256 + threadIdx.x;     // over NTOK*DINNER
    int d = idx & (DINNER - 1);
    int t = idx >> 10;                            // token = b*SEQ + l
    int l = t & (SEQ - 1);
    float acc = cb[d];
#pragma unroll
    for (int k = 0; k < DCONV; ++k) {
        int li = l - (DCONV - 1) + k;
        if (li >= 0)
            acc += cw[d * DCONV + k] *
                   xz[(size_t)(t - (DCONV - 1) + k) * (2 * DINNER) + d];
    }
    float sv = acc / (1.0f + __expf(-acc));       // SiLU
    xconv[idx]   = sv;
    xconv_h[idx] = (_Float16)sv;
}

// ------------------------------------------------- extract dt_low (f16) from x_dbl
__global__ __launch_bounds__(256) void k_extract_dt(const float* __restrict__ xdbl,
                                                    _Float16* __restrict__ dtlow) {
    int i = blockIdx.x * 256 + threadIdx.x;       // NTOK*DTRANK
    int r = i >> 5;
    int c = i & 31;
    dtlow[i] = (_Float16)xdbl[(size_t)r * 64 + c];
}

// ------------------------------------------------- selective scan
// One lane per (b, d, s): h kept in a register; 16-lane shfl_xor reduction for y.
// Fuses +x_conv*D and *silu(z), writes f16 input for out_proj GEMM.
__global__ __launch_bounds__(256) void k_scan(const float* __restrict__ dt,
                                              const float* __restrict__ xconv,
                                              const float* __restrict__ xdbl,
                                              const float* __restrict__ xz,
                                              const float* __restrict__ A_log,
                                              const float* __restrict__ Dp,
                                              _Float16* __restrict__ y_h) {
    int g = blockIdx.x * 256 + threadIdx.x;       // BATCH*DINNER*DSTATE lanes
    int s = g & 15;
    int d = (g >> 4) & (DINNER - 1);
    int b = g >> 14;
    float As = -__expf(A_log[d * DSTATE + s]);
    float Dv = Dp[d];
    float h = 0.f;
    size_t tok0 = (size_t)b * SEQ;
    for (int l = 0; l < SEQ; ++l) {
        size_t t = tok0 + l;
        float dtv = dt[t * DINNER + d];
        float xc  = xconv[t * DINNER + d];
        float Bv  = xdbl[t * 64 + DTRANK + s];
        float Cv  = xdbl[t * 64 + DTRANK + DSTATE + s];
        h = h * __expf(dtv * As) + dtv * Bv * xc;
        float c = h * Cv;
        c += __shfl_xor(c, 8, 16);
        c += __shfl_xor(c, 4, 16);
        c += __shfl_xor(c, 2, 16);
        c += __shfl_xor(c, 1, 16);
        if (s == 0) {
            float zv = xz[t * (2 * DINNER) + DINNER + d];
            float yv = (c + xc * Dv) * (zv / (1.0f + __expf(-zv)));
            y_h[t * DINNER + d] = (_Float16)yv;
        }
    }
}

// ---------------------------------------------------------------- launch
extern "C" void kernel_launch(void* const* d_in, const int* in_sizes, int n_in,
                              void* d_out, int out_size, void* d_ws, size_t ws_size,
                              hipStream_t stream) {
    const float* x         = (const float*)d_in[0];
    const float* ln_gamma  = (const float*)d_in[1];
    const float* ln_beta   = (const float*)d_in[2];
    const float* in_proj_w = (const float*)d_in[3];
    const float* conv_w    = (const float*)d_in[4];
    const float* conv_b    = (const float*)d_in[5];
    const float* x_proj_w  = (const float*)d_in[6];
    const float* dt_proj_w = (const float*)d_in[7];
    const float* dt_proj_b = (const float*)d_in[8];
    const float* A_log     = (const float*)d_in[9];
    const float* D_param   = (const float*)d_in[10];
    const float* out_proj_w= (const float*)d_in[11];
    float* out = (float*)d_out;

    char* ws = (char*)d_ws;
    size_t off = 0;
    auto alloc = [&](size_t bytes) {
        char* p = ws + off;
        off += (bytes + 255) & ~(size_t)255;
        return p;
    };
    _Float16* xnorm_h = (_Float16*)alloc((size_t)NTOK * DMODEL * 2);
    _Float16* wih     = (_Float16*)alloc((size_t)2 * DINNER * DMODEL * 2);
    _Float16* wxp     = (_Float16*)alloc((size_t)64 * DINNER * 2);
    _Float16* wdt     = (_Float16*)alloc((size_t)DINNER * DTRANK * 2);
    _Float16* wout    = (_Float16*)alloc((size_t)DMODEL * DINNER * 2);
    float*    xzbuf   = (float*)   alloc((size_t)NTOK * 2 * DINNER * 4);
    float*    xconv   = (float*)   alloc((size_t)NTOK * DINNER * 4);
    _Float16* xconv_h = (_Float16*)alloc((size_t)NTOK * DINNER * 2);
    float*    xdbl    = (float*)   alloc((size_t)NTOK * 64 * 4);
    _Float16* dtlow_h = (_Float16*)alloc((size_t)NTOK * DTRANK * 2);
    float*    dtbuf   = (float*)   alloc((size_t)NTOK * DINNER * 4);
    _Float16* y_h     = (_Float16*)alloc((size_t)NTOK * DINNER * 2);

    // weight fp32 -> f16 conversions (cheap, deterministic each launch)
    { int n = 2 * DINNER * DMODEL; k_f32_to_f16<<<(n + 255) / 256, 256, 0, stream>>>(in_proj_w, wih, n); }
    { int n = 64 * DINNER;         k_f32_to_f16<<<(n + 255) / 256, 256, 0, stream>>>(x_proj_w, wxp, n); }
    { int n = DINNER * DTRANK;     k_f32_to_f16<<<(n + 255) / 256, 256, 0, stream>>>(dt_proj_w, wdt, n); }
    { int n = DMODEL * DINNER;     k_f32_to_f16<<<(n + 255) / 256, 256, 0, stream>>>(out_proj_w, wout, n); }

    // layernorm -> f16 activations
    k_layernorm<<<NTOK / 8, 256, 0, stream>>>(x, ln_gamma, ln_beta, xnorm_h);

    // GEMM1: xz = x_norm @ in_proj_w^T      (8192 x 512 x 2048)
    { dim3 g(2 * DINNER / 64, NTOK / 128);
      k_wmma_gemm<<<g, 256, 0, stream>>>(xnorm_h, wih, xzbuf, NTOK, 2 * DINNER, DMODEL, 0, nullptr); }

    // depthwise causal conv + SiLU
    { int n = NTOK * DINNER;
      k_conv_silu<<<n / 256, 256, 0, stream>>>(xzbuf, conv_w, conv_b, xconv, xconv_h); }

    // GEMM2: x_dbl = x_conv @ x_proj_w^T    (8192 x 1024 x 64)
    { dim3 g(64 / 64, NTOK / 128);
      k_wmma_gemm<<<g, 256, 0, stream>>>(xconv_h, wxp, xdbl, NTOK, 64, DINNER, 0, nullptr); }

    // extract dt_low to f16
    { int n = NTOK * DTRANK;
      k_extract_dt<<<n / 256, 256, 0, stream>>>(xdbl, dtlow_h); }

    // GEMM3: dt = softplus(dt_low @ dt_proj_w^T + b)   (8192 x 32 x 1024)
    { dim3 g(DINNER / 64, NTOK / 128);
      k_wmma_gemm<<<g, 256, 0, stream>>>(dtlow_h, wdt, dtbuf, NTOK, DINNER, DTRANK, 1, dt_proj_b); }

    // selective scan (65536 lanes, state-parallel)
    { int n = BATCH * DINNER * DSTATE;
      k_scan<<<n / 256, 256, 0, stream>>>(dtbuf, xconv, xdbl, xzbuf, A_log, D_param, y_h); }

    // GEMM4: out = y @ out_proj_w^T         (8192 x 1024 x 512)
    { dim3 g(DMODEL / 64, NTOK / 128);
      k_wmma_gemm<<<g, 256, 0, stream>>>(y_h, wout, out, NTOK, DMODEL, DINNER, 0, nullptr); }
}
